// CrfRnnLayer_64604898066458
// MI455X (gfx1250) — matrix operand order repaired and checked by reading, not compiled, not run on hardware
//
#include <hip/hip_runtime.h>
#include <hip/hip_bf16.h>

// ---------------------------------------------------------------------------
// CRF-RNN mean-field layer, fused dense-bilateral via CDNA5 WMMA (gfx1250).
// H=96, W=128, C=21, N=12288, 5 iterations.
// bil = K @ sm^T with K = exp(-0.5*max(d2,0)) over 5-D features, fused:
// K tiles built in registers (VALU dot5 + v_exp_f32) feed
// v_wmma_f32_16x16x32_f16. Staging of softmax/feature chunks into LDS is
// done by the Tensor Data Mover (tensor_load_to_lds + s_wait_tensorcnt),
// double-buffered so the DMA hides behind WMMA/VALU work.
// ---------------------------------------------------------------------------

#define H_ 96
#define W_ 128
#define C_ 21
#define N_ (H_ * W_)          // 12288
#define R_ 9                   // 3-sigma radius, theta_gamma = 3
#define NUM_ITERS_ 5
#define JB_ 128                // j super-chunk staged per barrier (4 WMMA sub-chunks)

typedef __attribute__((ext_vector_type(16))) _Float16 v16h;
typedef __attribute__((ext_vector_type(8)))  _Float16 v8h;
typedef __attribute__((ext_vector_type(8)))  float    v8f;
typedef unsigned long long u64_t;
typedef __attribute__((ext_vector_type(4))) unsigned int u32x4;
typedef __attribute__((ext_vector_type(8))) int          i32x8;
typedef __attribute__((ext_vector_type(4))) int          i32x4;

#ifndef USE_TDM
#if defined(__has_builtin)
#if __has_builtin(__builtin_amdgcn_tensor_load_to_lds)
#define USE_TDM 1
#endif
#endif
#endif
#ifndef USE_TDM
#define USE_TDM 0
#endif

__device__ __forceinline__ float gweight(int d) {
    // exp(-0.5 * (d/3)^2)
    return __expf(-0.5f * (float)(d * d) * (1.0f / 9.0f));
}

#if USE_TDM
// Build a D# (groups 0/1 per CDNA5 ISA) for a 2-D
// tile (tile_d0 x tile_d1 elements) of a row-major tensor with row stride
// stride0 (elements), and issue TENSOR_LOAD_TO_LDS. data_size_log2: 0=1B,1=2B,2=4B.
// This toolchain exposes the 6-arg builtin form:
//   (u32x4 g0, i32x8 g1, i32x4 g2, i32x4 g3, i32x8 extra, i32 cpol)
// groups 2/3 and the extra group are zero (2-D tensor, no iterate/gather).
__device__ __forceinline__ void tdm_load_2d(unsigned lds_addr, const void* gaddr,
                                            unsigned tile_d0, unsigned tile_d1,
                                            unsigned tensor_d0, unsigned tensor_d1,
                                            unsigned long long stride0,
                                            int data_size_log2) {
    unsigned long long ga = (unsigned long long)gaddr;
    u32x4 g0;
    g0[0] = 1u;                                         // count=1, user mode
    g0[1] = lds_addr;                                   // LDS byte address
    g0[2] = (unsigned)(ga & 0xFFFFFFFFu);               // global_addr[31:0]
    g0[3] = (unsigned)((ga >> 32) & 0x01FFFFFFu)        // global_addr[56:32]
          | (2u << 30);                                 // type=2 ("image")
    i32x8 g1;
    g1[0] = (int)((unsigned)data_size_log2 << 16);      // wg_mask=0 | data_size
    g1[1] = (int)((tensor_d0 & 0xFFFFu) << 16);         // bar_addr=0 | tdim0[15:0]
    g1[2] = (int)(((tensor_d0 >> 16) & 0xFFFFu)
          |       ((tensor_d1 & 0xFFFFu) << 16));       // tdim0[31:16] | tdim1[15:0]
    g1[3] = (int)(((tensor_d1 >> 16) & 0xFFFFu)
          |       ((tile_d0 & 0xFFFFu) << 16));         // tdim1[31:16] | tile_dim0
    g1[4] = (int)(tile_d1 & 0xFFFFu);                   // tile_dim1 | tile_dim2=0
    g1[5] = (int)(unsigned)(stride0 & 0xFFFFFFFFu);     // dim0_stride[31:0]
    g1[6] = (int)(unsigned)((stride0 >> 32) & 0xFFFFu); // dim0_stride[47:32] | dim1_stride lo
    g1[7] = 0;
    i32x4 z4 = {0, 0, 0, 0};
    i32x8 z8 = {0, 0, 0, 0, 0, 0, 0, 0};
    __builtin_amdgcn_tensor_load_to_lds(g0, g1, z4, z4, z8, 0);
}
#endif

// ---------------------------------------------------------------------------
// Features: f[n] = { y/160, x/160, r/3, g/3, b/3, |f|^2, 0, 0 }
// ---------------------------------------------------------------------------
__global__ void k_features(const float* __restrict__ rgb, float* __restrict__ f) {
    int n = blockIdx.x * 256 + threadIdx.x;
    if (n >= N_) return;
    float fy = (float)(n / W_) * (1.0f / 160.0f);
    float fx = (float)(n % W_) * (1.0f / 160.0f);
    float fr = rgb[n * 3 + 0] * (1.0f / 3.0f);
    float fg = rgb[n * 3 + 1] * (1.0f / 3.0f);
    float fb = rgb[n * 3 + 2] * (1.0f / 3.0f);
    float sq = fy * fy + fx * fx + fr * fr + fg * fg + fb * fb;
    float* o = f + n * 8;
    o[0] = fy; o[1] = fx; o[2] = fr; o[3] = fg; o[4] = fb; o[5] = sq;
    o[6] = 0.0f; o[7] = 0.0f;
}

// ---------------------------------------------------------------------------
// q init: q[c][n] = unaries[n*C + c]   (HWC -> C-major)
// ---------------------------------------------------------------------------
__global__ void k_initq(const float* __restrict__ u, float* __restrict__ q) {
    int idx = blockIdx.x * 256 + threadIdx.x;
    if (idx >= C_ * N_) return;
    int c = idx / N_, n = idx % N_;
    q[c * N_ + n] = u[n * C_ + c];
}

// ---------------------------------------------------------------------------
// Softmax over channels; writes f32 (for spatial blur) and f16 in c-major
// [32][N] layout for WMMA B staging. Column 21 = 1.0 -> bil_norm for free.
// ---------------------------------------------------------------------------
__global__ void k_softmax(const float* __restrict__ q,
                          float* __restrict__ sm32,
                          _Float16* __restrict__ sm16) {
    int n = blockIdx.x * 256 + threadIdx.x;
    if (n >= N_) return;
    float v[C_];
    float m = -3.4e38f;
    #pragma unroll
    for (int c = 0; c < C_; ++c) { v[c] = q[c * N_ + n]; m = fmaxf(m, v[c]); }
    float s = 0.0f;
    #pragma unroll
    for (int c = 0; c < C_; ++c) { v[c] = __expf(v[c] - m); s += v[c]; }
    float inv = 1.0f / s;
    #pragma unroll
    for (int c = 0; c < C_; ++c) {
        float sm = v[c] * inv;
        sm32[c * N_ + n] = sm;
        sm16[c * N_ + n] = (_Float16)sm;
    }
    sm16[21 * N_ + n] = (_Float16)1.0f;          // ones column -> K @ 1 = bil_norm
    #pragma unroll
    for (int c = 22; c < 32; ++c) sm16[c * N_ + n] = (_Float16)0.0f;
}

// ---------------------------------------------------------------------------
// Separable Gaussian blur (zero-padded SAME), vertical then horizontal.
// ---------------------------------------------------------------------------
__global__ void k_blur_v(const float* __restrict__ in, float* __restrict__ out) {
    int idx = blockIdx.x * 256 + threadIdx.x;
    if (idx >= C_ * N_) return;
    int c = idx / N_, n = idx % N_;
    int y = n / W_, x = n % W_;
    float s = 0.0f;
    for (int d = -R_; d <= R_; ++d) {
        int yy = y + d;
        if ((unsigned)yy < (unsigned)H_)
            s = fmaf(gweight(d), in[c * N_ + yy * W_ + x], s);
    }
    out[idx] = s;
}

__global__ void k_blur_h(const float* __restrict__ in, float* __restrict__ out) {
    int idx = blockIdx.x * 256 + threadIdx.x;
    if (idx >= C_ * N_) return;
    int c = idx / N_, n = idx % N_;
    int y = n / W_, x = n % W_;
    float s = 0.0f;
    for (int d = -R_; d <= R_; ++d) {
        int xx = x + d;
        if ((unsigned)xx < (unsigned)W_)
            s = fmaf(gweight(d), in[c * N_ + y * W_ + xx], s);
    }
    out[idx] = s;
}

// ---------------------------------------------------------------------------
// Fused dense bilateral: bil[N][32] = exp-kernel-tile x sm^T via WMMA f16.
// 256 threads = 8 waves; wave w owns i-tile (blockIdx.x*8 + w) of 16 rows.
// Double-buffered TDM pipeline: wave 0 issues tensor_load_to_lds for super-
// chunk k+1 (softmax chunk [32c][128j] f16 + feature slab [128j][8] f32),
// all waves compute 4 WMMA sub-chunks from super-chunk k, wave 0 drains
// TENSORcnt, block barrier, repeat.
// ---------------------------------------------------------------------------
__global__ void __launch_bounds__(256)
k_bilateral(const float* __restrict__ f,
            const _Float16* __restrict__ sm16,
            float* __restrict__ bil) {
    __shared__ __align__(16) _Float16 sB[2][32 * JB_]; // [c][j] chunks of sm16
    __shared__ __align__(16) float    sF[2][JB_ * 8];  // [j][8] feature chunks

    const int tid  = threadIdx.x;
    const int lane = tid & 31;
    const int wave = tid >> 5;
    const int half = lane >> 4;     // 0: lanes 0-15, 1: lanes 16-31
    const int ln   = lane & 15;
    const int itile = blockIdx.x * 8 + wave;
    const int irow  = itile * 16 + ln;

    // This lane's feature vector (A-matrix row), constant over the j loop.
    const float* fi = f + irow * 8;
    const float fi0 = fi[0], fi1 = fi[1], fi2 = fi[2], fi3 = fi[3], fi4 = fi[4];
    const float fisq = fi[5];

    v8f acc0 = {};  // columns 0..15
    v8f acc1 = {};  // columns 16..31 (col 21 = bil_norm)

    const int nchunks = N_ / JB_;   // 96

#if USE_TDM
    // ---- TDM staging path -------------------------------------------------
    if (wave == 0) {
        tdm_load_2d((unsigned)(u64_t)(void*)&sB[0][0], sm16,
                    JB_, 32, N_, 32, (u64_t)N_, /*2B*/1);
        tdm_load_2d((unsigned)(u64_t)(void*)&sF[0][0], f,
                    JB_ * 8, 1, N_ * 8, 1, (u64_t)(N_ * 8), /*4B*/2);
        __builtin_amdgcn_s_wait_tensorcnt(0);
    }
    __syncthreads();
#else
    // ---- Manual staging fallback -----------------------------------------
    {
        const int r = tid >> 3, p = tid & 7;
        #pragma unroll
        for (int qd = 0; qd < 4; ++qd)
            ((u64_t*)&sB[0][r * JB_])[p + qd * 8] =
                ((const u64_t*)(sm16 + r * N_))[p + qd * 8];
        #pragma unroll
        for (int qd = 0; qd < 16; ++qd)
            sF[0][tid + qd * 256] = f[tid + qd * 256];
    }
    __syncthreads();
#endif

    for (int ch = 0; ch < nchunks; ++ch) {
        const int cur = ch & 1;
        const int jb_next = (ch + 1) * JB_;

#if USE_TDM
        if (wave == 0 && ch + 1 < nchunks) {
            tdm_load_2d((unsigned)(u64_t)(void*)&sB[cur ^ 1][0], sm16 + jb_next,
                        JB_, 32, N_, 32, (u64_t)N_, /*2B*/1);
            tdm_load_2d((unsigned)(u64_t)(void*)&sF[cur ^ 1][0], f + jb_next * 8,
                        JB_ * 8, 1, N_ * 8, 1, (u64_t)(N_ * 8), /*4B*/2);
        }
#endif

        // ---- 4 WMMA sub-chunks from buffer `cur` --------------------------
        #pragma unroll
        for (int s = 0; s < JB_ / 32; ++s) {
            // A fragment: 16x32 f16 tile of K, ISA A-matrix layout.
            v16h a;
            #pragma unroll
            for (int t = 0; t < 16; ++t) {
                const int k = s * 32 + ((t < 8) ? (t + 8 * half)
                                                : (t + 8 + 8 * half));
                const float* fr = &sF[cur][k * 8];
                float d2 = fisq + fr[5]
                         - 2.0f * (fi0 * fr[0] + fi1 * fr[1] + fi2 * fr[2]
                                 + fi3 * fr[3] + fi4 * fr[4]);
                d2 = fmaxf(d2, 0.0f);
                a[t] = (_Float16)__expf(-0.5f * d2);
            }
            // B fragments: lanes 0-15 K=0..7,16..23, 16-31 K=8..15,24..31, col = lane&15.
            const v8h* p0 = (const v8h*)&sB[cur][(ln)      * JB_ + s * 32];
            const v8h* p1 = (const v8h*)&sB[cur][(16 + ln) * JB_ + s * 32];
            v8h b0lo = p0[half], b0hi = p0[2 + half];
            v8h b1lo = p1[half], b1hi = p1[2 + half];
            v16h b0, b1;
            #pragma unroll
            for (int t = 0; t < 8; ++t) {
                b0[t] = b0lo[t]; b0[t + 8] = b0hi[t];
                b1[t] = b1lo[t]; b1[t + 8] = b1hi[t];
            }
            acc0 = __builtin_amdgcn_wmma_f32_16x16x32_f16(
                       false, a, false, b0, (short)0, acc0, false, false);
            acc1 = __builtin_amdgcn_wmma_f32_16x16x32_f16(
                       false, a, false, b1, (short)0, acc1, false, false);
        }

#if USE_TDM
        if (wave == 0) __builtin_amdgcn_s_wait_tensorcnt(0);
        __syncthreads();
#else
        if (ch + 1 < nchunks) {
            __syncthreads();
            const int r = tid >> 3, p = tid & 7;
            #pragma unroll
            for (int qd = 0; qd < 4; ++qd)
                ((u64_t*)&sB[cur ^ 1][r * JB_])[p + qd * 8] =
                    ((const u64_t*)(sm16 + r * N_ + jb_next))[p + qd * 8];
            #pragma unroll
            for (int qd = 0; qd < 16; ++qd)
                sF[cur ^ 1][tid + qd * 256] = f[jb_next * 8 + tid + qd * 256];
            __syncthreads();
        }
#endif
    }

    // Store D tile: element v of acc -> row M = v + 8*half, col = lane&15 (+16).
    #pragma unroll
    for (int v = 0; v < 8; ++v) {
        const int row = itile * 16 + v + 8 * half;
        bil[row * 32 + ln]      = acc0[v];
        bil[row * 32 + 16 + ln] = acc1[v];
    }
}

// ---------------------------------------------------------------------------
// Combine: q = u - comp @ (wsp @ (sp/sp_norm) + wbil @ (bil/bil_norm)).
// sp_norm(y,x) = S_H(y)*S_W(x) with truncated-tap sums (blur of ones).
// ---------------------------------------------------------------------------
__global__ void k_combine(const float* __restrict__ u,
                          const float* __restrict__ sp,
                          const float* __restrict__ bil,
                          const float* __restrict__ wsp,
                          const float* __restrict__ wbil,
                          const float* __restrict__ comp,
                          float* __restrict__ q) {
    int n = blockIdx.x * 256 + threadIdx.x;
    if (n >= N_) return;
    int y = n / W_, x = n % W_;
    float sy = 0.0f, sx = 0.0f;
    for (int d = -R_; d <= R_; ++d) {
        float g = gweight(d);
        if ((unsigned)(y + d) < (unsigned)H_) sy += g;
        if ((unsigned)(x + d) < (unsigned)W_) sx += g;
    }
    const float inv_sp = 1.0f / (sy * sx);
    const float inv_b  = 1.0f / bil[n * 32 + 21];

    float spn[C_], bln[C_], mp[C_];
    #pragma unroll
    for (int c = 0; c < C_; ++c) {
        spn[c] = sp[c * N_ + n] * inv_sp;
        bln[c] = bil[n * 32 + c] * inv_b;
    }
    for (int c = 0; c < C_; ++c) {
        float s = 0.0f;
        for (int k = 0; k < C_; ++k) {
            s = fmaf(wsp[c * C_ + k],  spn[k], s);
            s = fmaf(wbil[c * C_ + k], bln[k], s);
        }
        mp[c] = s;
    }
    for (int c = 0; c < C_; ++c) {
        float s = 0.0f;
        for (int k = 0; k < C_; ++k) s = fmaf(comp[c * C_ + k], mp[k], s);
        q[c * N_ + n] = u[n * C_ + c] - s;
    }
}

// ---------------------------------------------------------------------------
// Final transpose: out[n*C + c] = q[c][n]   (C-major -> HWC)
// ---------------------------------------------------------------------------
__global__ void k_transpose_out(const float* __restrict__ q, float* __restrict__ out) {
    int idx = blockIdx.x * 256 + threadIdx.x;
    if (idx >= C_ * N_) return;
    int c = idx / N_, n = idx % N_;
    out[n * C_ + c] = q[c * N_ + n];
}

// ---------------------------------------------------------------------------
extern "C" void kernel_launch(void* const* d_in, const int* in_sizes, int n_in,
                              void* d_out, int out_size, void* d_ws, size_t ws_size,
                              hipStream_t stream) {
    const float* unaries = (const float*)d_in[0];   // [1,H,W,C]
    const float* rgb     = (const float*)d_in[1];   // [1,H,W,3]
    const float* wsp     = (const float*)d_in[2];   // [C,C]
    const float* wbil    = (const float*)d_in[3];   // [C,C]
    const float* comp    = (const float*)d_in[4];   // [C,C]
    float* out = (float*)d_out;

    float* f    = (float*)d_ws;        // N*8
    float* q    = f    + N_ * 8;       // C*N
    float* sm32 = q    + C_ * N_;      // C*N
    float* tmp  = sm32 + C_ * N_;      // C*N
    float* sp   = tmp  + C_ * N_;      // C*N
    float* bil  = sp   + C_ * N_;      // N*32
    _Float16* sm16 = (_Float16*)(bil + N_ * 32);    // 32*N halves

    const int gN  = N_ / 256;                 // 48
    const int gCN = (C_ * N_ + 255) / 256;    // 1008

    k_features<<<gN, 256, 0, stream>>>(rgb, f);
    k_initq<<<gCN, 256, 0, stream>>>(unaries, q);

    for (int it = 0; it < NUM_ITERS_; ++it) {
        k_softmax<<<gN, 256, 0, stream>>>(q, sm32, sm16);
        k_blur_v<<<gCN, 256, 0, stream>>>(sm32, tmp);
        k_blur_h<<<gCN, 256, 0, stream>>>(tmp, sp);
        k_bilateral<<<(N_ / 16) / 8, 256, 0, stream>>>(f, sm16, bil); // 96 blocks
        k_combine<<<gN, 256, 0, stream>>>(unaries, sp, bil, wsp, wbil, comp, q);
    }
    k_transpose_out<<<gCN, 256, 0, stream>>>(q, out);
}
